// S4Convolution_84104049590477
// MI455X (gfx1250) — matrix operand order repaired and checked by reading, not compiled, not run on hardware
//
#include <hip/hip_runtime.h>

// S4 DPLR kernel + full correlation as a Toeplitz WMMA GEMM on gfx1250.
//   Stage 1: Cauchy dot products -> K_hat (L complex)          [trivial]
//   Stage 2: inverse DFT -> K (L real f32)                      [trivial]
//   Stage 3: out(4096 x 8191) = u(4096x4096) @ T(4096x8191),
//            T[m][j] = K[(L-1)+m-j], via v_wmma_f32_16x16x32_f16.
//
// GEMM tiling: block = 64(M) x 256(N), 8 waves, each wave owns a 16x128
// strip (8 WMMA C-tiles, 64 accumulator VGPRs). K stepped by 32.
// Pipelining:
//   - global f32 u slab prefetched into regs one k-step ahead (cvt->LDS),
//   - A tile double-buffered in LDS -> ONE barrier per k-step,
//   - B fragments double-buffered in VGPRs -> no full dscnt drain per WMMA.

#define S4_N    64
#define S4_L    4096
#define S4_JOUT 8191
#define S4_STEP 0.1f
#define PI_F    3.14159265358979323846f

typedef __attribute__((ext_vector_type(4)))  _Float16 v4h;
typedef __attribute__((ext_vector_type(8)))  _Float16 v8h;
typedef __attribute__((ext_vector_type(16))) _Float16 v16h;
typedef __attribute__((ext_vector_type(8)))  float    v8f;

// ---------------- Stage 1: generating function at roots of unity ----------
__global__ void s4_cauchy(const float* __restrict__ Lr, const float* __restrict__ Li,
                          const float* __restrict__ Pr, const float* __restrict__ Pi,
                          const float* __restrict__ Qr, const float* __restrict__ Qi,
                          const float* __restrict__ Br, const float* __restrict__ Bi,
                          const float* __restrict__ Cr, const float* __restrict__ Ci,
                          float2* __restrict__ Khat) {
  int k = blockIdx.x * blockDim.x + threadIdx.x;
  if (k >= S4_L) return;
  float ang = (2.0f * PI_F / (float)S4_L) * (float)k;
  float si, co;
  __sincosf(ang, &si, &co);
  float br = S4_STEP * (1.0f + co), bi = S4_STEP * si;      // b = STEP*(1+w)
  float d0r = 2.0f * (1.0f - co),  d0i = -2.0f * si;        // 2*(1-w)
  float s00r=0.f,s00i=0.f, s01r=0.f,s01i=0.f, s10r=0.f,s10i=0.f, s11r=0.f,s11i=0.f;
  for (int n = 0; n < S4_N; ++n) {
    float lr = Lr[n], li = Li[n];
    float dr = d0r - (br*lr - bi*li);
    float di = d0i - (br*li + bi*lr);
    float inv = 1.0f / (dr*dr + di*di);
    float rr = dr*inv, ri = -di*inv;                        // r = 1/denom
    float brn = Br[n], bin = Bi[n], prn = Pr[n], pin = Pi[n];
    float crn = Cr[n], cin = Ci[n], qrn = Qr[n], qin = Qi[n];
    float cbr = crn*brn + cin*bin, cbi = crn*bin - cin*brn; // conj(C)*B
    float cpr = crn*prn + cin*pin, cpi = crn*pin - cin*prn; // conj(C)*P
    float qbr = qrn*brn + qin*bin, qbi = qrn*bin - qin*brn; // conj(Q)*B
    float qpr = qrn*prn + qin*pin, qpi = qrn*pin - qin*prn; // conj(Q)*P
    s00r += rr*cbr - ri*cbi; s00i += rr*cbi + ri*cbr;
    s01r += rr*cpr - ri*cpi; s01i += rr*cpi + ri*cpr;
    s10r += rr*qbr - ri*qbi; s10i += rr*qbi + ri*qbr;
    s11r += rr*qpr - ri*qpi; s11i += rr*qpi + ri*qpr;
  }
  float tr = S4_STEP * br, ti = S4_STEP * bi;               // STEP*b
  float ar = tr*s01r - ti*s01i, ai = tr*s01i + ti*s01r;     // t*s01
  float nr = ar*s10r - ai*s10i, ni = ar*s10i + ai*s10r;     // *s10
  float er = 1.0f + tr*s11r - ti*s11i, ei = tr*s11i + ti*s11r; // 1 + t*s11
  float dinv = 1.0f / (er*er + ei*ei);
  float fr = (nr*er + ni*ei) * dinv, fi = (ni*er - nr*ei) * dinv;
  float2 kh;
  kh.x = 2.0f * S4_STEP * (s00r - fr);
  kh.y = 2.0f * S4_STEP * (s00i - fi);
  Khat[k] = kh;
}

// ---------------- Stage 2: inverse DFT (real part) -------------------------
__global__ void s4_idft(const float2* __restrict__ Khat, float* __restrict__ K) {
  int l = blockIdx.x * blockDim.x + threadIdx.x;
  if (l >= S4_L) return;
  const float c0 = 2.0f * PI_F / (float)S4_L;
  float acc = 0.0f;
  for (int k = 0; k < S4_L; ++k) {
    int m = (k * l) & (S4_L - 1);
    float si, co;
    __sincosf(c0 * (float)m, &si, &co);
    float2 h = Khat[k];
    acc += h.x * co - h.y * si;            // Re(K_hat[k] * e^{+2pi i k l / L})
  }
  K[l] = acc * (1.0f / (float)S4_L);
}

// ---------------- Stage 3: Toeplitz GEMM with WMMA --------------------------
// B operand is implicit Toeplitz: B[k][j] = K[(L-1)+k-j]. Per block we need
// W[s] = Kval(s + L-256 - j0), s in [0, 4352). Per-lane B-fragment reads are
// 16 consecutive halves starting at s0 = k + 255 - jj (2B-aligned only), so
// we keep 4 shifted copies Wc[r][i] = W[i+r]: with r = s0 & 3 every read is
// four aligned 8-byte chunks (fused into ds_load_2addr_b64 pairs).
#define S4_NT 256                 // block N tile
#define S4_LW (S4_L + S4_NT)      // 4352 window halves

__global__ __launch_bounds__(256) void s4_toeplitz_wmma(
    const float* __restrict__ u, const float* __restrict__ Kf,
    float* __restrict__ out) {
  __shared__ __align__(16) _Float16 Wc[4][S4_LW];    // 34816 B
  __shared__ __align__(16) _Float16 At[2][64 * 40];  // 2 x 5120 B double buffer

  const int tid = threadIdx.x;
  const int j0  = blockIdx.x * S4_NT;
  const int b0  = blockIdx.y * 64;

  // Build 4 shifted copies of the K window for this j-block (once).
  for (int r = 0; r < 4; ++r) {
    for (int i = tid; i < S4_LW; i += 256) {
      int ks = i + r + (S4_L - S4_NT) - j0;         // K index for W[i+r]
      float v = (ks >= 0 && ks < S4_L) ? Kf[ks] : 0.0f;
      Wc[r][i] = (_Float16)v;
    }
  }

  const int lane = tid & 31;
  const int w    = tid >> 5;
  const int wm   = w & 3;        // 0..3 -> 16-row M subtile
  const int wn   = w >> 2;       // 0..1 -> 128-col N group (8 tiles of 16)
  const int lh   = lane & 15;
  const int sel  = lane >> 4;    // half-wave select
  const int m    = wm * 16 + lh; // A-matrix row for this lane

  v8f acc[8] = {};

  // Loop-invariant fragment addressing.
  const int afo = m * 40 + sel * 16;                // A frag offset (halves)
  const _Float16* bp[8];                            // B frag bases at k0 = 0
  #pragma unroll
  for (int t = 0; t < 8; ++t) {
    const int jj = wn * 128 + t * 16 + lh;          // block-local column
    const int s0 = sel * 16 + (S4_NT - 1) - jj;     // window index at k = sel*16
    const int r  = s0 & 3;                          // k-invariant copy select
    bp[t] = &Wc[r][s0 - r];                         // 8B-aligned; advance +k0
  }

  // Cooperative A staging: 64 rows x 32 halves, 8 halves per thread.
  const int arow = tid >> 2;
  const int acol = (tid & 3) * 8;
  const int aso  = arow * 40 + acol;                // A store offset (halves)
  const float* usrc = u + (size_t)(b0 + arow) * S4_L + acol;

  auto cvt8 = [](float4 f0, float4 f1) -> v8h {
    v8h hv;
    hv[0]=(_Float16)f0.x; hv[1]=(_Float16)f0.y; hv[2]=(_Float16)f0.z; hv[3]=(_Float16)f0.w;
    hv[4]=(_Float16)f1.x; hv[5]=(_Float16)f1.y; hv[6]=(_Float16)f1.z; hv[7]=(_Float16)f1.w;
    return hv;
  };
  auto loadB = [&](int t, int k0) -> v16h {
    const _Float16* wp = bp[t] + k0;                // +32 halves per k-step
    union { v16h v; v4h q[4]; } bu;
    bu.q[0] = *(const v4h*)(wp + 0);
    bu.q[1] = *(const v4h*)(wp + 4);
    bu.q[2] = *(const v4h*)(wp + 8);
    bu.q[3] = *(const v4h*)(wp + 12);
    return bu.v;
  };

  // One k-step: read A from `cur`, stage slab (k0+32) into `nxt`.
  auto step = [&](int k0, const _Float16* cur, _Float16* nxt) {
    const bool have_next = (k0 + 32) < S4_L;
    float4 f0, f1;
    if (have_next) {
      f0 = *(const float4*)(usrc + k0 + 32);
      f1 = *(const float4*)(usrc + k0 + 36);
      if (k0 + 64 < S4_L) __builtin_prefetch(usrc + k0 + 64, 0, 3);
    }

    // A fragment (16x32 f16): lane<16 -> K 0..15, lane>=16 -> 16..31.
    union { v16h v; v8h h[2]; } au;
    au.h[0] = *(const v8h*)(cur + afo);
    au.h[1] = *(const v8h*)(cur + afo + 8);

    // B fragments double-buffered in VGPRs: load t+1 under WMMA t.
    v16h bcur = loadB(0, k0), bnxt;
    #pragma unroll
    for (int t = 0; t < 8; ++t) {
      if (t < 7) bnxt = loadB(t + 1, k0);
      acc[t] = __builtin_amdgcn_wmma_f32_16x16x32_f16(
          false, au.v, false, bcur, (short)0, acc[t], false, false);
      bcur = bnxt;
    }

    if (have_next) *(v8h*)(nxt + aso) = cvt8(f0, f1);
  };

  // Prologue: stage slab k0 = 0 into buffer 0; publish Wc + At[0].
  {
    float4 f0 = *(const float4*)(usrc);
    float4 f1 = *(const float4*)(usrc + 4);
    *(v8h*)(&At[0][0] + aso) = cvt8(f0, f1);
  }
  __syncthreads();

  // 128 k-steps, unrolled by 2 for static buffer parity; ONE barrier/step.
  for (int k0 = 0; k0 < S4_L; k0 += 64) {
    step(k0,      &At[0][0], &At[1][0]);
    __syncthreads();
    step(k0 + 32, &At[1][0], &At[0][0]);
    __syncthreads();
  }

  // C/D layout: VGPR r -> row (wm*16 + sel*8 + r), cols = low-16 lanes.
  #pragma unroll
  for (int t = 0; t < 8; ++t) {
    int col = j0 + wn * 128 + t * 16 + lh;
    if (col < S4_JOUT) {
      float* op = out + (size_t)(b0 + wm * 16 + sel * 8) * S4_JOUT + col;
      #pragma unroll
      for (int r = 0; r < 8; ++r) op[(size_t)r * S4_JOUT] = acc[t][r];
    }
  }
}

// ---------------------------------------------------------------------------
extern "C" void kernel_launch(void* const* d_in, const int* in_sizes, int n_in,
                              void* d_out, int out_size, void* d_ws, size_t ws_size,
                              hipStream_t stream) {
  (void)in_sizes; (void)n_in; (void)out_size; (void)ws_size;
  const float* u  = (const float*)d_in[0];
  const float* Lr = (const float*)d_in[1];
  const float* Li = (const float*)d_in[2];
  const float* Pr = (const float*)d_in[3];
  const float* Pi = (const float*)d_in[4];
  const float* Qr = (const float*)d_in[5];
  const float* Qi = (const float*)d_in[6];
  const float* Br = (const float*)d_in[7];
  const float* Bi = (const float*)d_in[8];
  const float* Cr = (const float*)d_in[9];
  const float* Ci = (const float*)d_in[10];

  float2* Khat = (float2*)d_ws;                                   // 32 KB
  float*  Kf   = (float*)((char*)d_ws + S4_L * sizeof(float2));   // 16 KB

  s4_cauchy<<<S4_L / 256, 256, 0, stream>>>(Lr, Li, Pr, Pi, Qr, Qi,
                                            Br, Bi, Cr, Ci, Khat);
  s4_idft<<<S4_L / 256, 256, 0, stream>>>(Khat, Kf);

  dim3 grid(S4_JOUT / S4_NT + 1, 64);  // 32 j-tiles x 64 batch-tiles
  s4_toeplitz_wmma<<<grid, 256, 0, stream>>>(u, Kf, (float*)d_out);
}
